// TernaryCIFAR10Net_31164282699839
// MI455X (gfx1250) — matrix operand order repaired and checked by reading, not compiled, not run on hardware
//
#include <hip/hip_runtime.h>
#include <hip/hip_bf16.h>

// ---------------------------------------------------------------------------
// MI455X (gfx1250) ternary CIFAR net. All matrix math via v_wmma_f32_16x16x32_f16.
// Conv = implicit GEMM with im2col tile staged in LDS; bias+ReLU+2x2 maxpool
// fused into the conv epilogue (pool windows never straddle a block's pixel
// tile), so no full-resolution activations ever touch HBM.
//
// d_ws layout (_Float16 elements), total ~62 MB:
//   A1   @ 0         : 16,777,216  pooled act1 [1024,64,16,16]
//   A2   @ 16777216  :  8,388,608  pooled act2 [1024,128,8,8]
//   A3   @ 25165824  :  4,194,304  pooled act3 [1024,256,4,4] (fc1 input)
//   A4   @ 29360128  :    262,144  fc1 output  [1024,256]
//   W1T..WF2T        : ternarized f16 weight matrices, padded [OCpad][Kt]
// ---------------------------------------------------------------------------

typedef __attribute__((ext_vector_type(16))) _Float16 v16h;
typedef __attribute__((ext_vector_type(8)))  _Float16 v8h;
typedef __attribute__((ext_vector_type(8)))  float    v8f;

union V16 { v16h v; v8h h[2]; _Float16 e[16]; };
union V8  { v8h  v; _Float16 e[8]; };

// ---------------------------------------------------------------------------
// Ternarize one weight tensor (TWN) and write padded f16 A-matrix [Mpad][Kt].
// Single workgroup of 1024 threads; deterministic reduction order.
// ---------------------------------------------------------------------------
__global__ void ternarize_pad_kernel(const float* __restrict__ w,
                                     _Float16* __restrict__ out,
                                     int M, int K, int Mpad, int Kt) {
  __shared__ float sred[1024];
  __shared__ float sval[2];
  const int tid = threadIdx.x;
  const int S = M * K;

  // pass 1: sum |w|  -> delta = 0.7 * mean|w|
  float loc = 0.f;
  for (int i = tid; i < S; i += 1024) loc += fabsf(w[i]);
  sred[tid] = loc; __syncthreads();
  for (int s = 512; s > 0; s >>= 1) {
    if (tid < s) sred[tid] += sred[tid + s];
    __syncthreads();
  }
  if (tid == 0) sval[0] = 0.7f * sred[0] / (float)S;
  __syncthreads();
  const float delta = sval[0];

  // pass 2: alpha = sum(|w| * mask) / max(count, 1)
  float la = 0.f, lc = 0.f;
  for (int i = tid; i < S; i += 1024) {
    float a = fabsf(w[i]);
    if (a > delta) { la += a; lc += 1.f; }
  }
  __syncthreads();
  sred[tid] = la; __syncthreads();
  for (int s = 512; s > 0; s >>= 1) {
    if (tid < s) sred[tid] += sred[tid + s];
    __syncthreads();
  }
  if (tid == 0) sval[0] = sred[0];
  __syncthreads();
  const float sumA = sval[0];
  __syncthreads();
  sred[tid] = lc; __syncthreads();
  for (int s = 512; s > 0; s >>= 1) {
    if (tid < s) sred[tid] += sred[tid + s];
    __syncthreads();
  }
  if (tid == 0) sval[1] = sred[0];
  __syncthreads();
  const float alpha = sumA / fmaxf(sval[1], 1.f);

  // pass 3: write padded ternary f16 matrix
  const int tot = Mpad * Kt;
  for (int i = tid; i < tot; i += 1024) {
    int m = i / Kt, k = i % Kt;
    float v = 0.f;
    if (m < M && k < K) {
      float x = w[(size_t)m * K + k];
      if (fabsf(x) > delta) v = (x > 0.f) ? alpha : -alpha;
    }
    out[i] = (_Float16)v;
  }
}

// ---------------------------------------------------------------------------
// Implicit-GEMM 3x3 conv (pad 1) + bias + ReLU + fused 2x2 maxpool.
// GEMM: C[oc][pixel] = sum_k Wt[oc][k] * im2col[k][pixel], k = cin*9 + r*3 + s
// Block: 256 threads = 8 waves = 2 OC-subtiles x 4 pixel-subtiles
//        -> 32 OC x 64 pixels per block.  K-loop in steps of 32, im2col tile
//        staged in LDS each step.  The 64-pixel tile always covers an even
//        number of complete, even-aligned rows of one image, so 2x2 pooling
//        is resolved in-block through LDS; only pooled f16 goes to HBM.
// ---------------------------------------------------------------------------
template <bool INF32>
__global__ void conv_implicit_wmma(const void* __restrict__ xin,
                                   const _Float16* __restrict__ Wt,
                                   const float* __restrict__ bias,
                                   _Float16* __restrict__ ypool,
                                   int Cin, int H, int Wd, int OC, int Kt) {
  __shared__ _Float16 Bt[64 * 32];  // im2col tile [px][k]; reused as relu tile [px][oc]

  const int tid  = threadIdx.x;
  const int lane = tid & 31;
  const int wave = tid >> 5;
  const int ocSub = wave >> 2;      // 0..1
  const int pxSub = wave & 3;       // 0..3
  const int h2    = lane >> 4;      // 0..1
  const int HW    = H * Wd;
  const int Kreal = Cin * 9;

  const int ocRow = blockIdx.y * 32 + ocSub * 16 + (lane & 15);  // A row (M)

  // staging coords: thread t loads 8 consecutive k for pixel pr, octet q
  const int pr = tid >> 2;
  const int q  = tid & 3;
  const long p_st = (long)blockIdx.x * 64 + pr;
  const int b_st  = (int)(p_st / HW);
  const int hw_st = (int)(p_st % HW);
  const int hh_st = hw_st / Wd;
  const int ww_st = hw_st % Wd;

  v8f acc = {};

  for (int k0 = 0; k0 < Kt; k0 += 32) {
    // ---- stage im2col tile 64 x 32 into LDS ----
    V8 stg;
#pragma unroll
    for (int i = 0; i < 8; ++i) {
      int k = k0 + q * 8 + i;
      float v = 0.f;
      if (k < Kreal) {
        int cin = k / 9, rs = k % 9;
        int hy = hh_st + rs / 3 - 1;
        int wx = ww_st + rs % 3 - 1;
        if (hy >= 0 && hy < H && wx >= 0 && wx < Wd) {
          size_t idx = ((size_t)(b_st * Cin + cin) * H + hy) * Wd + wx;
          v = INF32 ? ((const float*)xin)[idx]
                    : (float)((const _Float16*)xin)[idx];
        }
      }
      stg.e[i] = (_Float16)v;
    }
    *(v8h*)(&Bt[pr * 32 + q * 8]) = stg.v;
    __syncthreads();

    // ---- fragments + WMMA ----
    // A (16-bit A 16x32 layout): lane m = lane%16, K = {8*h2+0..7, 16+8*h2+0..7}
    const _Float16* ap = Wt + (size_t)ocRow * Kt + k0;
    if (k0 + 32 < Kt) __builtin_prefetch(ap + 32, 0, 1);
    V16 a;
    a.h[0] = *(const v8h*)(ap + 8 * h2);
    a.h[1] = *(const v8h*)(ap + 16 + 8 * h2);
    // B (16-bit B 32x16 layout): lane n = lane%16, K = {8*h2+0..7, 16+8*h2+0..7}
    const int nloc = pxSub * 16 + (lane & 15);
    V16 bv;
    bv.h[0] = *(const v8h*)(&Bt[nloc * 32 + 8 * h2]);
    bv.h[1] = *(const v8h*)(&Bt[nloc * 32 + 16 + 8 * h2]);

    acc = __builtin_amdgcn_wmma_f32_16x16x32_f16(
        false, a.v, false, bv.v, (short)0, acc, false, false);
    __syncthreads();
  }

  // ---- epilogue 1: bias + ReLU -> LDS relu tile [px 0..63][oc 0..31] ----
  {
    const int nloc = pxSub * 16 + (lane & 15);
    const int ocBaseL = ocSub * 16 + 8 * h2;   // local oc of acc[0]
    V8 st;
#pragma unroll
    for (int i = 0; i < 8; ++i) {
      int oc = blockIdx.y * 32 + ocBaseL + i;  // D: M = 8*h2 + i
      st.e[i] = (_Float16)fmaxf(acc[i] + bias[oc], 0.f);
    }
    *(v8h*)(&Bt[nloc * 32 + ocBaseL]) = st.v;
  }
  __syncthreads();

  // ---- epilogue 2: 2x2 maxpool inside the block, write pooled f16 ----
  // Block covers rows h0 .. h0 + 64/Wd - 1 (h0 even) of image b.
  const long p0 = (long)blockIdx.x * 64;
  const int b  = (int)(p0 / HW);
  const int h0 = (int)((p0 % HW) / Wd);
  const int Wo = Wd >> 1;
  const int Ho = H >> 1;
#pragma unroll
  for (int j = 0; j < 2; ++j) {
    int idx  = tid + 256 * j;      // 0..511 = 32 oc x 16 pooled px
    int ocL  = idx >> 4;           // 0..31
    int ppx  = idx & 15;           // 0..15
    int prow = ppx / Wo;
    int pcol = ppx % Wo;
    int p00  = (2 * prow) * Wd + 2 * pcol;   // local pixel of top-left
    float m0 = fmaxf((float)Bt[p00 * 32 + ocL],
                     (float)Bt[(p00 + 1) * 32 + ocL]);
    float m1 = fmaxf((float)Bt[(p00 + Wd) * 32 + ocL],
                     (float)Bt[(p00 + Wd + 1) * 32 + ocL]);
    int oc = blockIdx.y * 32 + ocL;
    int ho = (h0 >> 1) + prow;
    ypool[((size_t)(b * OC + oc) * Ho + ho) * Wo + pcol] = (_Float16)fmaxf(m0, m1);
  }
}

// ---------------------------------------------------------------------------
// FC wave-GEMM: out[n][m] = sum_k A[m][k] * Bact[n][k]  (+bias, opt ReLU)
// mode 0: ReLU -> f16 outH[n*M + m]      (fc1)
// mode 1: f32  -> outF[n*biasN + m], m < biasN   (fc2, M padded to 16)
// ---------------------------------------------------------------------------
__global__ void fc_wmma_kernel(const _Float16* __restrict__ A,
                               const _Float16* __restrict__ Bact,
                               const float* __restrict__ bias,
                               _Float16* __restrict__ outH,
                               float* __restrict__ outF,
                               int M, int K, int biasN, int mode) {
  const int tid  = threadIdx.x;
  const int lane = tid & 31;
  const int wave = tid >> 5;
  const int mSub = wave >> 2;  // 0..1
  const int nSub = wave & 3;   // 0..3
  const int mBase = blockIdx.y * 32 + mSub * 16;
  if (mBase >= M) return;      // wave-uniform (fc2 uses only 4 of 8 waves)
  const int h2 = lane >> 4;
  const int mRow = mBase + (lane & 15);
  const long n = (long)blockIdx.x * 64 + nSub * 16 + (lane & 15);

  v8f acc = {};
  for (int k0 = 0; k0 < K; k0 += 32) {
    const _Float16* ap = A + (size_t)mRow * K + k0;
    V16 a;
    a.h[0] = *(const v8h*)(ap + 8 * h2);
    a.h[1] = *(const v8h*)(ap + 16 + 8 * h2);
    const _Float16* bp = Bact + (size_t)n * K + k0 + 8 * h2;
    V16 bv;
    bv.h[0] = *(const v8h*)(bp);
    bv.h[1] = *(const v8h*)(bp + 16);
    acc = __builtin_amdgcn_wmma_f32_16x16x32_f16(
        false, a.v, false, bv.v, (short)0, acc, false, false);
  }

#pragma unroll
  for (int i = 0; i < 8; ++i) {
    int m = mBase + 8 * h2 + i;
    float v = acc[i] + ((m < biasN) ? bias[m] : 0.f);
    if (mode == 0) {
      v = fmaxf(v, 0.f);
      outH[(size_t)n * M + m] = (_Float16)v;
    } else if (m < biasN) {
      outF[(size_t)n * biasN + m] = v;
    }
  }
}

// ---------------------------------------------------------------------------
extern "C" void kernel_launch(void* const* d_in, const int* in_sizes, int n_in,
                              void* d_out, int out_size, void* d_ws, size_t ws_size,
                              hipStream_t stream) {
  (void)in_sizes; (void)n_in; (void)out_size; (void)ws_size;

  const float* x   = (const float*)d_in[0];   // [1024,3,32,32]
  const float* w1  = (const float*)d_in[1];   // [64,3,3,3]
  const float* b1  = (const float*)d_in[2];
  const float* w2  = (const float*)d_in[3];   // [128,64,3,3]
  const float* b2  = (const float*)d_in[4];
  const float* w3  = (const float*)d_in[5];   // [256,128,3,3]
  const float* b3  = (const float*)d_in[6];
  const float* wf1 = (const float*)d_in[7];   // [256,4096]
  const float* bf1 = (const float*)d_in[8];
  const float* wf2 = (const float*)d_in[9];   // [10,256]
  const float* bf2 = (const float*)d_in[10];
  float* out = (float*)d_out;                 // [1024,10] f32

  _Float16* ws = (_Float16*)d_ws;
  const size_t A1   = 0;                      // 16,777,216 halfs
  const size_t A2   = A1   + 16777216ull;     //  8,388,608
  const size_t A3   = A2   +  8388608ull;     //  4,194,304
  const size_t A4   = A3   +  4194304ull;     //    262,144
  const size_t W1T  = A4   +   262144ull;     //      2,048  (64 x 32)
  const size_t W2T  = W1T  +     2048ull;     //     73,728  (128 x 576)
  const size_t W3T  = W2T  +    73728ull;     //    294,912  (256 x 1152)
  const size_t WF1T = W3T  +   294912ull;     //  1,048,576  (256 x 4096)
  const size_t WF2T = WF1T +  1048576ull;     //      4,096  (16 x 256)

  // --- ternarize all weight tensors into padded f16 A-matrices ---
  ternarize_pad_kernel<<<1, 1024, 0, stream>>>(w1,  ws + W1T,  64,   27,  64,   32);
  ternarize_pad_kernel<<<1, 1024, 0, stream>>>(w2,  ws + W2T, 128,  576, 128,  576);
  ternarize_pad_kernel<<<1, 1024, 0, stream>>>(w3,  ws + W3T, 256, 1152, 256, 1152);
  ternarize_pad_kernel<<<1, 1024, 0, stream>>>(wf1, ws + WF1T,256, 4096, 256, 4096);
  ternarize_pad_kernel<<<1, 1024, 0, stream>>>(wf2, ws + WF2T, 10,  256,  16,  256);

  // --- conv1 + relu + pool: [1024,3,32,32] -> A1 [1024,64,16,16] ---
  conv_implicit_wmma<true><<<dim3(16384, 2), 256, 0, stream>>>(
      x, ws + W1T, b1, ws + A1, 3, 32, 32, 64, 32);

  // --- conv2 + relu + pool: A1 -> A2 [1024,128,8,8] ---
  conv_implicit_wmma<false><<<dim3(4096, 4), 256, 0, stream>>>(
      ws + A1, ws + W2T, b2, ws + A2, 64, 16, 16, 128, 576);

  // --- conv3 + relu + pool: A2 -> A3 [1024,256,4,4] ---
  conv_implicit_wmma<false><<<dim3(1024, 8), 256, 0, stream>>>(
      ws + A2, ws + W3T, b3, ws + A3, 128, 8, 8, 256, 1152);

  // --- fc1: [1024,4096] x [256,4096]^T + bias, ReLU -> A4 (f16 [1024,256]) ---
  fc_wmma_kernel<<<dim3(16, 8), 256, 0, stream>>>(
      ws + WF1T, ws + A3, bf1, ws + A4, nullptr, 256, 4096, 256, 0);

  // --- fc2: [1024,256] x [16,256]^T + bias -> d_out (f32 [1024,10]) ---
  fc_wmma_kernel<<<dim3(16, 1), 256, 0, stream>>>(
      ws + WF2T, ws + A4, bf2, nullptr, out, 16, 256, 10, 1);
}
